// AutoCorrelation_17136919511813
// MI455X (gfx1250) — compile-verified, matrix operand order checked by reading, not run on hardware
//
#include <hip/hip_runtime.h>
#include <hip/hip_bf16.h>

// Problem constants (from reference): B=8, N=128, L=192, H=8, E=D=64
#define B_     8
#define N_     128
#define L_     192
#define CH_    512              // H*E == H*D flattened channel dim
#define TOPK   5                // int(log(192)) == 5
#define CHUNK  128              // channels per LDS phase
#define NPHASE (CH_ / CHUNK)    // 4
#define STRIDE 136              // padded LDS row stride (bf16 elems): 272B -> conflict-free
#define NT     (L_ / 16)        // 12 tiles per time dim

typedef __attribute__((ext_vector_type(16))) __bf16 v16bf;
typedef __attribute__((ext_vector_type(8)))  float  v8f;

__device__ __forceinline__ unsigned short f2bf(float f) {
  union { float f; unsigned int u; } x; x.f = f;
  unsigned int r = x.u + 0x7FFFu + ((x.u >> 16) & 1u);   // round-to-nearest-even
  return (unsigned short)(r >> 16);
}

// ---------------------------------------------------------------------------
// Kernel 1: mean_value[b,n,tau] = (1/512) * sum_t sum_c q[t,c] * k[(t+tau)%L,c]
//           == summed diagonals of C = Q * K^T   (Q,K are [192 x 512] per b,n)
// One workgroup per (b,n). bf16 WMMA, fp32 accumulate, LDS diagonal reduction.
// ---------------------------------------------------------------------------
__global__ __launch_bounds__(256)
void corr_kernel(const float* __restrict__ q, const float* __restrict__ k,
                 float* __restrict__ mv) {
  __shared__ unsigned short qb[L_ * STRIDE];
  __shared__ unsigned short kb[L_ * STRIDE];
  __shared__ float msum[L_];

  const int bn   = blockIdx.x;
  const size_t base = (size_t)bn * L_ * CH_;
  const int tid  = threadIdx.x;
  const int lane = tid & 31;
  const int wave = tid >> 5;
  const int half = (lane >> 4) & 1;   // K-half select per WMMA 16-bit layout
  const int r    = lane & 15;         // row (A) / col (B) within 16x16 tile

  if (tid < L_) msum[tid] = 0.0f;
  __syncthreads();

  for (int ph = 0; ph < NPHASE; ++ph) {
    // ---- stage CHUNK channels of q,k into LDS as bf16 (float4-vectorized) ----
    const int co = ph * CHUNK;
    for (int i = tid; i < L_ * (CHUNK / 4); i += 256) {
      const int row = i >> 5;          // 32 float4 per row
      const int c4  = i & 31;
      const float4 qa = *(const float4*)(q + base + (size_t)row * CH_ + co + c4 * 4);
      const float4 ka = *(const float4*)(k + base + (size_t)row * CH_ + co + c4 * 4);
      const unsigned int q01 = (unsigned int)f2bf(qa.x) | ((unsigned int)f2bf(qa.y) << 16);
      const unsigned int q23 = (unsigned int)f2bf(qa.z) | ((unsigned int)f2bf(qa.w) << 16);
      const unsigned int k01 = (unsigned int)f2bf(ka.x) | ((unsigned int)f2bf(ka.y) << 16);
      const unsigned int k23 = (unsigned int)f2bf(ka.z) | ((unsigned int)f2bf(ka.w) << 16);
      *(uint2*)&qb[row * STRIDE + c4 * 4] = make_uint2(q01, q23);
      *(uint2*)&kb[row * STRIDE + c4 * 4] = make_uint2(k01, k23);
    }
    __syncthreads();

    // ---- each wave sweeps its share of the 12x12 grid of 16x16 C tiles ----
    for (int tile = wave; tile < NT * NT; tile += 8) {
      const int ti = tile / NT;        // row-block of Q (time t)
      const int tj = tile % NT;        // row-block of K (time t')
      v8f acc = {0.f, 0.f, 0.f, 0.f, 0.f, 0.f, 0.f, 0.f};
      const unsigned short* arow = &qb[(ti * 16 + r) * STRIDE];
      const unsigned short* brow = &kb[(tj * 16 + r) * STRIDE];
#pragma unroll
      for (int kk = 0; kk < CHUNK / 32; ++kk) {
        union { v16bf v; uint4 u[2]; } fa, fb;
        const int c0 = kk * 32 + half * 8;       // 16B aligned in LDS
        const int c1 = kk * 32 + 16 + half * 8;
        fa.u[0] = *(const uint4*)(arow + c0);
        fa.u[1] = *(const uint4*)(arow + c1);
        fb.u[0] = *(const uint4*)(brow + c0);
        fb.u[1] = *(const uint4*)(brow + c1);
        acc = __builtin_amdgcn_wmma_f32_16x16x32_bf16(false, fa.v, false, fb.v,
                                                      (short)0, acc, false, false);
      }
      // ---- diagonal reduction: C[t,t'] -> msum[(t'-t) mod L] ----
      const int Mbase = ti * 16 + (half ? 8 : 0);  // C/D layout: lanes>=16 hold M+8
      const int Ng    = tj * 16 + r;               // N = lane % 16
#pragma unroll
      for (int g = 0; g < 8; ++g) {
        int tau = Ng - (Mbase + g);
        if (tau < 0) tau += L_;
        atomicAdd(&msum[tau], acc[g]);
      }
    }
    __syncthreads();
  }

  if (tid < L_) mv[(size_t)bn * L_ + tid] = msum[tid] * (1.0f / (float)CH_);
}

// ---------------------------------------------------------------------------
// Kernel 2: g[n,tau] = sum_b mean_value[b,n,tau]; idx[n,0..4] = top-5 taus
// ---------------------------------------------------------------------------
__global__ void topk_kernel(const float* __restrict__ mv, int* __restrict__ idx) {
  __shared__ float g[L_];
  const int n = blockIdx.x;
  const int tid = threadIdx.x;
  if (tid < L_) {
    float s = 0.f;
    for (int b = 0; b < B_; ++b) s += mv[((size_t)b * N_ + n) * L_ + tid];
    g[tid] = s;
  }
  __syncthreads();
  if (tid == 0) {
    for (int j = 0; j < TOPK; ++j) {    // K=5, serial argmax matches lax.top_k order
      int best = 0; float bv = g[0];
      for (int t = 1; t < L_; ++t) { if (g[t] > bv) { bv = g[t]; best = t; } }
      idx[n * TOPK + j] = best;
      g[best] = -3.4e38f;
    }
  }
}

// ---------------------------------------------------------------------------
// Kernel 3: out[b,n,t,:] = sum_j softmax_j(mv[b,n,idx[n,j]]) * v[b,n,(t+idx_j)%L,:]
// One block per (b,n, 16-row t-chunk).
// ---------------------------------------------------------------------------
__global__ __launch_bounds__(256)
void agg_kernel(const float* __restrict__ v, const float* __restrict__ mv,
                const int* __restrict__ idx, float* __restrict__ out) {
  __shared__ int   sidx[TOPK];
  __shared__ float sw[TOPK];
  const int blk = blockIdx.x;
  const int tc  = blk % NT;
  const int bn  = blk / NT;
  const int n   = bn % N_;
  const int tid = threadIdx.x;

  if (tid == 0) {
    float raw[TOPK]; int id[TOPK];
    float m = -3.4e38f;
    for (int j = 0; j < TOPK; ++j) {
      id[j]  = idx[n * TOPK + j];
      raw[j] = mv[(size_t)bn * L_ + id[j]];
      m = fmaxf(m, raw[j]);
    }
    float s = 0.f;
    for (int j = 0; j < TOPK; ++j) { raw[j] = __expf(raw[j] - m); s += raw[j]; }
    const float inv = 1.0f / s;
    for (int j = 0; j < TOPK; ++j) { sw[j] = raw[j] * inv; sidx[j] = id[j]; }
  }
  __syncthreads();

  const size_t base = (size_t)bn * L_ * CH_;
  for (int i = tid; i < 16 * (CH_ / 4); i += 256) {   // 2048 float4 per block
    const int row = tc * 16 + (i >> 7);               // CH_/4 == 128 float4/row
    const int c4  = i & 127;
    float4 acc = {0.f, 0.f, 0.f, 0.f};
#pragma unroll
    for (int j = 0; j < TOPK; ++j) {
      int rr = row + sidx[j];
      if (rr >= L_) rr -= L_;
      const float4 vv = *(const float4*)(v + base + (size_t)rr * CH_ + c4 * 4);
      const float w = sw[j];
      acc.x += w * vv.x; acc.y += w * vv.y; acc.z += w * vv.z; acc.w += w * vv.w;
    }
    *(float4*)(out + base + (size_t)row * CH_ + c4 * 4) = acc;
  }
}

extern "C" void kernel_launch(void* const* d_in, const int* in_sizes, int n_in,
                              void* d_out, int out_size, void* d_ws, size_t ws_size,
                              hipStream_t stream) {
  const float* q = (const float*)d_in[0];
  const float* k = (const float*)d_in[1];
  const float* v = (const float*)d_in[2];
  float* out = (float*)d_out;

  float* mv   = (float*)d_ws;                                   // [B,N,L]
  int*   idxp = (int*)((char*)d_ws + (size_t)B_ * N_ * L_ * sizeof(float));

  corr_kernel<<<B_ * N_,      256, 0, stream>>>(q, k, mv);
  topk_kernel<<<N_,           192, 0, stream>>>(mv, idxp);
  agg_kernel <<<B_ * N_ * NT, 256, 0, stream>>>(v, mv, idxp, out);
}
